// LESS4FD_20899310862415
// MI455X (gfx1250) — matrix operand order repaired and checked by reading, not compiled, not run on hardware
//
#include <hip/hip_runtime.h>

typedef __attribute__((ext_vector_type(16))) _Float16 v16h;
typedef __attribute__((ext_vector_type(8)))  float    v8f;

#define F_IN 512
#define HID  128
#define NCLS 16

// Permute K within a 32-chunk so each lane's A fragment is contiguous:
// K 0-7 -> 0-7, K 8-15 -> 16-23, K 16-23 -> 8-15, K 24-31 -> 24-31
__device__ __forceinline__ int posmap(int k) {
    return (((k >> 3) ^ (k >> 4)) & 1) ? (k ^ 24) : k;
}

// ---------------- weight packing (wave32 WMMA B-operand layout) ----------------
// B layout (16x16x32 f16): lane l, elem h holds K = 8*(l>>4) + h (h<8) or 16 + 8*(l>>4) + h-8 (h>=8), N = l&15
__global__ void pack_w1(const float* __restrict__ W1l, const float* __restrict__ W1g,
                        _Float16* __restrict__ W1p) {
    int s = blockIdx.x * 256 + threadIdx.x;            // 2*16*8*32 = 8192 slots
    if (s >= 2 * 16 * 8 * 32) return;
    int lane = s & 31, ct = (s >> 5) & 7, kc = (s >> 8) & 15, b = s >> 12;
    const float* W = b ? W1g : W1l;
    int Kbase = kc * 32 + ((lane & 16) ? 8 : 0);
    int Ncol  = ct * 16 + (lane & 15);
    _Float16* dst = W1p + (size_t)s * 16;
    for (int h = 0; h < 16; ++h)
        dst[h] = (_Float16)W[(size_t)(Kbase + h + (h >= 8 ? 8 : 0)) * HID + Ncol];
}

__global__ void pack_w2(const float* __restrict__ W2l, const float* __restrict__ W2g,
                        _Float16* __restrict__ W2p) {
    int s = threadIdx.x;                               // 2*4*32 = 256 slots
    int lane = s & 31, kc = (s >> 5) & 3, b = s >> 7;
    const float* W = b ? W2g : W2l;
    int Kbase = kc * 32 + ((lane & 16) ? 8 : 0);
    int Ncol  = lane & 15;
    _Float16* dst = W2p + (size_t)s * 16;
    for (int h = 0; h < 16; ++h)
        dst[h] = (_Float16)W[(size_t)(Kbase + h + (h >= 8 ? 8 : 0)) * NCLS + Ncol];
}

// ---------------- degree / norm ----------------
__global__ void deg_init(float* deg, int N) {
    int i = blockIdx.x * 256 + threadIdx.x;
    if (i < N) deg[i] = 1.0f;                          // self-loop
}
__global__ void deg_accum(const long long* __restrict__ ei, float* deg, int E) {
    int e = blockIdx.x * 256 + threadIdx.x;
    if (e < E) unsafeAtomicAdd(&deg[(int)ei[(size_t)E + e]], 1.0f);
}
__global__ void make_dinv(float* deg, int N) {
    int i = blockIdx.x * 256 + threadIdx.x;
    if (i < N) deg[i] = rsqrtf(deg[i]);                // deg >= 1 always
}
__global__ void make_norm(const long long* __restrict__ ei, const float* __restrict__ dinv,
                          float* __restrict__ normE, int E) {
    int e = blockIdx.x * 256 + threadIdx.x;
    if (e < E) normE[e] = dinv[(int)ei[e]] * dinv[(int)ei[(size_t)E + e]];
}

// ---------------- fused GEMM1 (both branches) + GEMM2 ----------------
__global__ void __launch_bounds__(256)
fused_gemm(const float* __restrict__ x,
           const _Float16* __restrict__ W1p,
           const float* __restrict__ b1l, const float* __restrict__ b1g,
           const _Float16* __restrict__ W2p,
           const float* __restrict__ b2l, const float* __restrict__ b2g,
           const float* __restrict__ templ, const float* __restrict__ tempg,
           float* __restrict__ out,
           float* __restrict__ curl, float* __restrict__ curg,
           float* __restrict__ hidl, float* __restrict__ hidg,
           int N) {
    __shared__ __align__(32) _Float16 ldsA[16 * 32];       // x tile, f16, K-permuted
    __shared__ __align__(32) _Float16 lds2[2 * 16 * HID];  // relu(h1) tiles, K-permuted

    const int tid  = threadIdx.x;
    const int lane = tid & 31;
    const int w    = tid >> 5;       // wave 0..7
    const int b    = w >> 2;         // branch
    const int p    = w & 3;          // column pair: cols [32p, 32p+32)
    const int g    = lane >> 4;
    const int ln   = lane & 15;
    const int row0 = blockIdx.x * 16;

    v8f acc0 = {}; v8f acc1 = {};

    for (int kc = 0; kc < 16; ++kc) {
        {   // cooperative stage: 16x32 fp32 tile -> f16 LDS (permuted)
            int idx = tid * 2;
            int r = idx >> 5;
            int k = idx & 31;
            int rr = row0 + r; if (rr >= N) rr = N - 1;
            const float* src = x + (size_t)rr * F_IN + kc * 32 + k;
            float2 v = *(const float2*)src;
            int pp = posmap(k);                      // even k: pair stays adjacent
            ldsA[r * 32 + pp]     = (_Float16)v.x;
            ldsA[r * 32 + pp + 1] = (_Float16)v.y;
        }
        __syncthreads();
        v16h a = *(const v16h*)&ldsA[ln * 32 + g * 16];
        const _Float16* wp = W1p + ((((size_t)b * 16 + kc) * 8 + (size_t)p * 2) * 32 + lane) * 16;
        v16h wb0 = *(const v16h*)wp;
        v16h wb1 = *(const v16h*)(wp + 32 * 16);
        acc0 = __builtin_amdgcn_wmma_f32_16x16x32_f16(false, a, false, wb0, (short)0, acc0, false, false);
        acc1 = __builtin_amdgcn_wmma_f32_16x16x32_f16(false, a, false, wb1, (short)0, acc1, false, false);
        __syncthreads();
    }

    // bias + relu, write h1, stage f16 h1 into LDS as GEMM2 A-operand
    const float* b1v = b ? b1g : b1l;
    const float bias0 = b1v[p * 32 + ln];
    const float bias1 = b1v[p * 32 + 16 + ln];
    const int   moff  = g * 8;
    const size_t branchOff = (size_t)b * (size_t)N * 160;
    float* h1out = out + branchOff;
    const int pos0 = (ln < 8) ? ln : ln + 8;          // posmap(n), n in 0..15
    const int n1   = 16 + ln;
    const int pos1 = (n1 < 24) ? n1 - 8 : n1;         // posmap(n), n in 16..31
#pragma unroll
    for (int r = 0; r < 8; ++r) {
        int M = r + moff;
        float v0 = fmaxf(acc0[r] + bias0, 0.0f);
        float v1 = fmaxf(acc1[r] + bias1, 0.0f);
        if (row0 + M < N) {
            h1out[(size_t)(row0 + M) * HID + p * 32 + ln]      = v0;
            h1out[(size_t)(row0 + M) * HID + p * 32 + 16 + ln] = v1;
        }
        lds2[(b * 16 + M) * HID + p * 32 + pos0] = (_Float16)v0;
        lds2[(b * 16 + M) * HID + p * 32 + pos1] = (_Float16)v1;
    }
    __syncthreads();

    // GEMM2: waves 0 (branch l) and 4 (branch g); K = 128 = 4 x 32
    if ((w & 3) == 0) {
        v8f acc2 = {};
#pragma unroll
        for (int kc2 = 0; kc2 < 4; ++kc2) {
            v16h a2 = *(const v16h*)&lds2[(b * 16 + ln) * HID + kc2 * 32 + g * 16];
            v16h wb = *(const v16h*)(W2p + (((size_t)b * 4 + kc2) * 32 + lane) * 16);
            acc2 = __builtin_amdgcn_wmma_f32_16x16x32_f16(false, a2, false, wb, (short)0, acc2, false, false);
        }
        const float bias2 = (b ? b2g : b2l)[ln];
        const float t0    = (b ? tempg : templ)[0];
        float* cur = b ? curg : curl;
        float* hid = b ? hidg : hidl;
#pragma unroll
        for (int r = 0; r < 8; ++r) {
            int M = r + moff;
            if (row0 + M < N) {
                float v = acc2[r] + bias2;
                size_t o = (size_t)(row0 + M) * NCLS + ln;
                cur[o] = v;            // z0
                hid[o] = t0 * v;       // temp[0] * z0
            }
        }
    }
}

// ---------------- propagation ----------------
// next = dinv^2 * cur (self-loop), fused with hid += temp[k-1]*cur for k>=2
__global__ void prop_init_acc(const float* __restrict__ dinv,
                              const float* __restrict__ templ, const float* __restrict__ tempg,
                              const float* __restrict__ curl, const float* __restrict__ curg,
                              float* __restrict__ nextl, float* __restrict__ nextg,
                              float* __restrict__ hidl, float* __restrict__ hidg,
                              int k, int N16) {
    int i = blockIdx.x * 256 + threadIdx.x;
    if (i >= N16) return;
    float d = dinv[i >> 4], d2 = d * d;
    float cl = curl[i], cg = curg[i];
    if (k > 1) {
        hidl[i] += templ[k - 1] * cl;
        hidg[i] += tempg[k - 1] * cg;
    }
    nextl[i] = d2 * cl;
    nextg[i] = d2 * cg;
}

__global__ void prop_scatter(const long long* __restrict__ ei, const float* __restrict__ normE,
                             const float* __restrict__ curl, const float* __restrict__ curg,
                             float* __restrict__ nextl, float* __restrict__ nextg, int E) {
    int t = blockIdx.x * 256 + threadIdx.x;
    int e = t >> 4, c = t & 15;
    if (e >= E) return;
    int row = (int)ei[e];
    int col = (int)ei[(size_t)E + e];
    float nv = normE[e];
    unsafeAtomicAdd(&nextl[(size_t)col * NCLS + c], nv * curl[(size_t)row * NCLS + c]);
    unsafeAtomicAdd(&nextg[(size_t)col * NCLS + c], nv * curg[(size_t)row * NCLS + c]);
}

__global__ void final_acc(const float* __restrict__ templ, const float* __restrict__ tempg,
                          const float* __restrict__ curl, const float* __restrict__ curg,
                          float* __restrict__ hidl, float* __restrict__ hidg, int K, int N16) {
    int i = blockIdx.x * 256 + threadIdx.x;
    if (i >= N16) return;
    hidl[i] += templ[K] * curl[i];
    hidg[i] += tempg[K] * curg[i];
}

__global__ void finalize(const float* __restrict__ hidl, const float* __restrict__ hidg,
                         float* __restrict__ out, int N) {
    int i = blockIdx.x * 256 + threadIdx.x;
    if (i >= N) return;
    for (int b = 0; b < 2; ++b) {
        const float* hid = b ? hidg : hidl;
        size_t bo = (size_t)b * (size_t)N * 160;
        float v[NCLS], m = -3.0e38f;
        for (int c = 0; c < NCLS; ++c) { v[c] = hid[(size_t)i * NCLS + c]; m = fmaxf(m, v[c]); }
        float s = 0.0f;
        for (int c = 0; c < NCLS; ++c) s += expf(v[c] - m);
        float l = logf(s);
        float* h2o = out + bo + (size_t)N * HID;
        float* pro = out + bo + (size_t)N * (HID + NCLS);
        for (int c = 0; c < NCLS; ++c) {
            h2o[(size_t)i * NCLS + c] = v[c];
            pro[(size_t)i * NCLS + c] = v[c] - m - l;
        }
    }
}

// ---------------- host ----------------
extern "C" void kernel_launch(void* const* d_in, const int* in_sizes, int n_in,
                              void* d_out, int out_size, void* d_ws, size_t ws_size,
                              hipStream_t stream) {
    const float*     x     = (const float*)d_in[0];
    const long long* ei    = (const long long*)d_in[1];
    const float*     W1l   = (const float*)d_in[2];
    const float*     b1l   = (const float*)d_in[3];
    const float*     W2l   = (const float*)d_in[4];
    const float*     b2l   = (const float*)d_in[5];
    const float*     templ = (const float*)d_in[6];
    const float*     W1g   = (const float*)d_in[7];
    const float*     b1g   = (const float*)d_in[8];
    const float*     W2g   = (const float*)d_in[9];
    const float*     b2g   = (const float*)d_in[10];
    const float*     tempg = (const float*)d_in[11];

    const int N  = in_sizes[0] / F_IN;
    const int E  = in_sizes[1] / 2;
    const int KH = in_sizes[6] - 1;
    float* out = (float*)d_out;

    // workspace carve-out (256B aligned slices)
    char* wsp = (char*)d_ws;
    auto carve = [&](size_t bytes) { char* p = wsp; wsp += (bytes + 255) & ~(size_t)255; return p; };
    float*    deg   = (float*)carve((size_t)N * 4);        // deg -> dinv (in place)
    float*    normE = (float*)carve((size_t)E * 4);
    size_t    fvb   = (size_t)N * NCLS * 4;
    float*    bufAl = (float*)carve(fvb);
    float*    bufBl = (float*)carve(fvb);
    float*    hidl  = (float*)carve(fvb);
    float*    bufAg = (float*)carve(fvb);
    float*    bufBg = (float*)carve(fvb);
    float*    hidg  = (float*)carve(fvb);
    _Float16* W1p   = (_Float16*)carve((size_t)2 * 16 * 8 * 32 * 16 * 2);
    _Float16* W2p   = (_Float16*)carve((size_t)2 * 4 * 32 * 16 * 2);

    pack_w1<<<(8192 + 255) / 256, 256, 0, stream>>>(W1l, W1g, W1p);
    pack_w2<<<1, 256, 0, stream>>>(W2l, W2g, W2p);

    deg_init <<<(N + 255) / 256, 256, 0, stream>>>(deg, N);
    deg_accum<<<(E + 255) / 256, 256, 0, stream>>>(ei, deg, E);
    make_dinv<<<(N + 255) / 256, 256, 0, stream>>>(deg, N);
    make_norm<<<(E + 255) / 256, 256, 0, stream>>>(ei, deg, normE, E);

    fused_gemm<<<(N + 15) / 16, 256, 0, stream>>>(x, W1p, b1l, b1g, W2p, b2l, b2g,
                                                  templ, tempg, out,
                                                  bufAl, bufAg, hidl, hidg, N);

    const int N16 = N * NCLS;
    const int gN16 = (N16 + 255) / 256;
    const int gE   = (int)(((long long)E * NCLS + 255) / 256);
    float *curl = bufAl, *nextl = bufBl, *curg = bufAg, *nextg = bufBg;
    for (int k = 1; k <= KH; ++k) {
        prop_init_acc<<<gN16, 256, 0, stream>>>(deg, templ, tempg, curl, curg,
                                                nextl, nextg, hidl, hidg, k, N16);
        prop_scatter<<<gE, 256, 0, stream>>>(ei, normE, curl, curg, nextl, nextg, E);
        float* t;
        t = curl; curl = nextl; nextl = t;
        t = curg; curg = nextg; nextg = t;
    }
    final_acc<<<gN16, 256, 0, stream>>>(templ, tempg, curl, curg, hidl, hidg, KH, N16);
    finalize <<<(N + 255) / 256, 256, 0, stream>>>(hidl, hidg, out, N);
}
